// TimeDomainRWKVSeparator_73684458930209
// MI455X (gfx1250) — compile-verified, matrix operand order checked by reading, not run on hardware
//
#include <hip/hip_runtime.h>
#include <hip/hip_bf16.h>
#include <math.h>

// ---------------------------------------------------------------- constants
#define N_EMBD     256
#define N_LAYER    2
#define NUM_SRC    2
#define HEAD_SZ    64
#define N_HEAD     4          // N_EMBD / HEAD_SZ
#define BATCH      2
#define SEQ        4096
#define BT         (BATCH * SEQ)     // 8192 token rows
#define D4         (4 * N_EMBD)      // 1024

typedef __attribute__((ext_vector_type(16))) _Float16 v16h;
typedef __attribute__((ext_vector_type(8)))  _Float16 v8h;
typedef __attribute__((ext_vector_type(8)))  float    v8f;

enum { EP_F32 = 0, EP_ADD = 1, EP_SIGMOID = 2, EP_DECAY = 3, EP_RELU2_H = 4 };

// ---------------------------------------------------------------- embed
// x[bt,d] = mix[bt] * in_w[d] + in_b[d]     (mix is [B,1,T] == [BT])
__global__ void k_embed(const float* __restrict__ mix, const float* __restrict__ in_w,
                        const float* __restrict__ in_b, float* __restrict__ x) {
    int idx = blockIdx.x * blockDim.x + threadIdx.x;
    if (idx >= BT * N_EMBD) return;
    int bt = idx >> 8, d = idx & 255;
    x[idx] = mix[bt] * in_w[d] + in_b[d];
}

// ---------------------------------------------------------------- layernorm
// One wave32 per row of 256; 8 floats per lane; butterfly reduce.
__global__ void __launch_bounds__(256)
k_layernorm(const float* __restrict__ x, const float* __restrict__ s,
            const float* __restrict__ bsh, float* __restrict__ out,
            _Float16* __restrict__ out_h) {
    int lane = threadIdx.x & 31;
    int wv   = threadIdx.x >> 5;
    int row  = blockIdx.x * (blockDim.x >> 5) + wv;
    if (row >= BT) return;
    const float* xr = x + (size_t)row * N_EMBD;
    float v[8];
    float sum = 0.f;
#pragma unroll
    for (int i = 0; i < 8; ++i) { v[i] = xr[i * 32 + lane]; sum += v[i]; }
#pragma unroll
    for (int m = 16; m >= 1; m >>= 1) sum += __shfl_xor(sum, m, 32);
    float mean = sum * (1.0f / N_EMBD);
    float s2 = 0.f;
#pragma unroll
    for (int i = 0; i < 8; ++i) { float dd = v[i] - mean; s2 += dd * dd; }
#pragma unroll
    for (int m = 16; m >= 1; m >>= 1) s2 += __shfl_xor(s2, m, 32);
    float rstd = rsqrtf(s2 * (1.0f / N_EMBD) + 1e-5f);
#pragma unroll
    for (int i = 0; i < 8; ++i) {
        int d = i * 32 + lane;
        float o = (v[i] - mean) * rstd * s[d] + bsh[d];
        out[(size_t)row * N_EMBD + d] = o;
        if (out_h) out_h[(size_t)row * N_EMBD + d] = (_Float16)o;
    }
}

// ---------------------------------------------------------------- token-shift mix
// out = f16( x + (shift(x)-x) * mu )
__global__ void k_mix(const float* __restrict__ xln, const float* __restrict__ mu,
                      _Float16* __restrict__ out) {
    int idx = blockIdx.x * blockDim.x + threadIdx.x;
    if (idx >= BT * N_EMBD) return;
    int bt = idx >> 8, d = idx & 255;
    int t = bt & (SEQ - 1);
    float cur  = xln[idx];
    float prev = (t == 0) ? 0.f : xln[idx - N_EMBD];
    out[idx] = (_Float16)(cur + (prev - cur) * mu[d]);
}

// ---------------------------------------------------------------- weight transpose+convert
// W[K,N] f32  ->  Wt[N,K] f16  (so WMMA B fragments are contiguous rows)
__global__ void k_transpose(const float* __restrict__ W, _Float16* __restrict__ Wt,
                            int K, int N) {
    int idx = blockIdx.x * blockDim.x + threadIdx.x;
    if (idx >= K * N) return;
    int k = idx / N, n = idx - k * N;
    Wt[(size_t)n * K + k] = (_Float16)W[idx];
}

// ---------------------------------------------------------------- WMMA fragment load
// 16-bit 16x32 fragment per CDNA5 layout: lane<16 -> K{k0..k0+7, k0+16..k0+23},
// lane>=16 -> K{k0+8..k0+15, k0+24..k0+31}. `p` is the per-lane base
// (row/col and sel*8 already applied); only +k0 varies inside the K loop.
__device__ __forceinline__ v16h load_frag16(const _Float16* __restrict__ p) {
    v8h lo = *(const v8h*)p;
    v8h hi = *(const v8h*)(p + 16);
    v16h o;
#pragma unroll
    for (int i = 0; i < 8; ++i) { o[i] = lo[i]; o[i + 8] = hi[i]; }
    return o;
}

// ---------------------------------------------------------------- WMMA GEMM + epilogues
// C[M,N] (+)= A[M,K]f16 @ Wt[N,K]^T f16.
// One 16(M) x 64(N) super-tile per wave: 4 named accumulators share one A
// fragment -> 4 independent back-to-back v_wmma issues per K-slice and 4x
// A reuse. Loop body is branch-free (clamped speculative prefetch) with all
// fragment loads grouped ahead of the WMMAs so the scheduler can pipeline
// with partial s_wait_loadcnt. 8 waves per block.
__global__ void __launch_bounds__(256)
k_wmma_gemm(const _Float16* __restrict__ A, const _Float16* __restrict__ Bt,
            float* __restrict__ C, _Float16* __restrict__ Ch,
            const float* __restrict__ bias, int M, int N, int K, int ep) {
    int lane = threadIdx.x & 31;
    int wave = blockIdx.x * (blockDim.x >> 5) + (threadIdx.x >> 5);
    int tn = N >> 6;                       // N in 64-wide super-tiles (all N % 64 == 0)
    int tiles = (M >> 4) * tn;
    if (wave >= tiles) return;             // uniform per wave (EXEC all-1 for WMMA)
    int tm_i = wave / tn, tn_i = wave - tm_i * tn;
    int m0 = tm_i << 4, n0 = tn_i << 6;

    int r = lane & 15, sel = lane >> 4;
    const _Float16* Ap = A  + (size_t)(m0 + r) * K + sel * 8;       // per-lane A base
    const _Float16* Bp = Bt + (size_t)(n0 + r) * K + sel * 8;       // per-lane B base (q=0)
    const size_t bstride = (size_t)16 * K;                           // q -> q+1

    v8f acc0 = {}, acc1 = {}, acc2 = {}, acc3 = {};

    for (int k0 = 0; k0 < K; k0 += 32) {
        v16h a  = load_frag16(Ap + k0);
        v16h b0 = load_frag16(Bp + k0);
        v16h b1 = load_frag16(Bp + bstride     + k0);
        v16h b2 = load_frag16(Bp + bstride * 2 + k0);
        v16h b3 = load_frag16(Bp + bstride * 3 + k0);
        int kp = k0 + 64; kp = (kp < K) ? kp : (K - 32);             // clamp, no branch
        __builtin_prefetch(Ap + kp, 0, 1);                           // global_prefetch_b8
        __builtin_prefetch(Bp + kp, 0, 1);
        acc0 = __builtin_amdgcn_wmma_f32_16x16x32_f16(false, a, false, b0, (short)0, acc0, false, false);
        acc1 = __builtin_amdgcn_wmma_f32_16x16x32_f16(false, a, false, b1, (short)0, acc1, false, false);
        acc2 = __builtin_amdgcn_wmma_f32_16x16x32_f16(false, a, false, b2, (short)0, acc2, false, false);
        acc3 = __builtin_amdgcn_wmma_f32_16x16x32_f16(false, a, false, b3, (short)0, acc3, false, false);
    }

#pragma unroll
    for (int q = 0; q < 4; ++q) {
        v8f accq = (q == 0) ? acc0 : (q == 1) ? acc1 : (q == 2) ? acc2 : acc3;
        int n = n0 + 16 * q + r;
        float bv = bias ? bias[n] : 0.f;
#pragma unroll
        for (int rr = 0; rr < 8; ++rr) {
            int m = m0 + rr + sel * 8;     // D layout: VGPR rr -> M=rr / M=rr+8
            size_t idx = (size_t)m * N + n;
            float vacc = accq[rr] + bv;
            if (ep == EP_F32)          C[idx] = vacc;
            else if (ep == EP_ADD)     C[idx] += vacc;
            else if (ep == EP_SIGMOID) C[idx] = 1.f / (1.f + __expf(-vacc));
            else if (ep == EP_DECAY) { // exp(-softplus(v)-0.5)
                float sp = (vacc > 20.f) ? vacc : log1pf(__expf(vacc));
                C[idx] = __expf(-sp - 0.5f);
            } else {                    // EP_RELU2_H -> f16 out
                float t = vacc > 0.f ? vacc : 0.f;
                Ch[idx] = (_Float16)(t * t);
            }
        }
    }
}

// ---------------------------------------------------------------- kk / removal / replace prep
// kk = k / (||k||+1e-6) per (bt,h); ah = -kk; bh = kk * a
__global__ void k_kkprep(const float* __restrict__ k, const float* __restrict__ a,
                         float* __restrict__ ah, float* __restrict__ bh) {
    int idx = blockIdx.x * blockDim.x + threadIdx.x;
    if (idx >= BT * N_HEAD) return;
    int bt = idx >> 2, h = idx & 3;
    size_t base = (size_t)bt * N_EMBD + h * HEAD_SZ;
    float ss = 0.f;
    for (int j = 0; j < HEAD_SZ; ++j) { float kv = k[base + j]; ss += kv * kv; }
    float inv = 1.f / (sqrtf(ss) + 1e-6f);
    for (int j = 0; j < HEAD_SZ; ++j) {
        float kk = k[base + j] * inv;
        ah[base + j] = -kk;
        bh[base + j] = kk * a[base + j];
    }
}

// ---------------------------------------------------------------- RWKV-7 sequential scan
// 1 block per (b,h); 64 threads; thread i owns state row S[i,0..63] in VGPRs.
// Double-buffered LDS: global loads for step t+1 issue before the compute of
// step t. Update loop reads one packed float4 (w, kk*a, k, r) broadcast per j.
__global__ void __launch_bounds__(64)
k_scan(const float* __restrict__ r, const float* __restrict__ w,
       const float* __restrict__ k, const float* __restrict__ v,
       const float* __restrict__ ah, const float* __restrict__ bh,
       float* __restrict__ y) {
    __shared__ float4 pk[2][64];   // (w, bh, k, r)
    __shared__ float  la[2][64];   // ah (removal dir)
    __shared__ float  lv[2][64];   // v
    int b = blockIdx.x >> 2;
    int h = blockIdx.x & 3;
    int i = threadIdx.x;
    float S[64];
#pragma unroll
    for (int j = 0; j < 64; ++j) S[j] = 0.f;

    size_t base0 = ((size_t)(b * SEQ)) * N_EMBD + h * HEAD_SZ + i;
    // stage step 0 into registers
    float sw = w[base0], sb = bh[base0], sk = k[base0], sr = r[base0];
    float sav = ah[base0], sv = v[base0];

    for (int t = 0; t < SEQ; ++t) {
        int buf = t & 1;
        pk[buf][i] = make_float4(sw, sb, sk, sr);
        la[buf][i] = sav;
        lv[buf][i] = sv;
        __syncthreads();
        if (t + 1 < SEQ) {          // issue next step's loads before compute
            size_t nb = base0 + (size_t)(t + 1) * N_EMBD;
            sw = w[nb]; sb = bh[nb]; sk = k[nb]; sr = r[nb];
            sav = ah[nb]; sv = v[nb];
        }
        float vi = lv[buf][i];
        float sa = 0.f;
#pragma unroll
        for (int j = 0; j < 64; ++j) sa += S[j] * la[buf][j];   // pre-update S
        float yi = 0.f;
#pragma unroll
        for (int j = 0; j < 64; ++j) {
            float4 c = pk[buf][j];
            S[j] = S[j] * c.x + sa * c.y + vi * c.z;
            yi  += S[j] * c.w;
        }
        y[base0 + (size_t)t * N_EMBD] = yi;
        __syncthreads();
    }
}

// ---------------------------------------------------------------- group-norm + gate -> f16
__global__ void k_gngate(const float* __restrict__ y, const float* __restrict__ g,
                         const float* __restrict__ gs, const float* __restrict__ gb,
                         _Float16* __restrict__ out) {
    int idx = blockIdx.x * blockDim.x + threadIdx.x;
    if (idx >= BT * N_HEAD) return;
    int bt = idx >> 2, h = idx & 3;
    size_t base = (size_t)bt * N_EMBD + h * HEAD_SZ;
    float sum = 0.f, sq = 0.f;
    for (int j = 0; j < HEAD_SZ; ++j) { float v = y[base + j]; sum += v; sq += v * v; }
    float m = sum * (1.f / HEAD_SZ);
    float var = sq * (1.f / HEAD_SZ) - m * m;
    float rstd = rsqrtf(var + 1e-5f);
    for (int j = 0; j < HEAD_SZ; ++j) {
        int d = h * HEAD_SZ + j;
        float xn = (y[base + j] - m) * rstd;
        out[base + j] = (_Float16)((xn * gs[d] + gb[d]) * g[base + j]);
    }
}

// ---------------------------------------------------------------- masked head -> [B,S,T]
__global__ void k_maskout(const float* __restrict__ mask, const float* __restrict__ h,
                          const float* __restrict__ ow, const float* __restrict__ ob,
                          float* __restrict__ out) {
    int idx = blockIdx.x * blockDim.x + threadIdx.x;
    if (idx >= BT * NUM_SRC) return;
    int bt = idx >> 1, s = idx & 1;
    float acc = 0.f;
    size_t mb = (size_t)bt * (NUM_SRC * N_EMBD) + s * N_EMBD;
    size_t hb = (size_t)bt * N_EMBD;
    for (int d = 0; d < N_EMBD; ++d) acc += mask[mb + d] * h[hb + d] * ow[d];
    int b = bt / SEQ, t = bt - b * SEQ;
    out[((size_t)b * NUM_SRC + s) * SEQ + t] = acc + ob[0];
}

// ---------------------------------------------------------------- host side
static inline int cdiv(int a, int b) { return (a + b - 1) / b; }

struct Bump {
    char* p; size_t off; size_t cap;
    void* alloc(size_t bytes) {
        off = (off + 255) & ~(size_t)255;
        void* r = p + off;
        off += bytes;
        return r;
    }
};

extern "C" void kernel_launch(void* const* d_in, const int* in_sizes, int n_in,
                              void* d_out, int out_size, void* d_ws, size_t ws_size,
                              hipStream_t stream) {
    (void)in_sizes; (void)n_in; (void)out_size;
    // Inputs in setup_inputs() insertion order.
    const float* mixin = (const float*)d_in[0];
    const float* in_w  = (const float*)d_in[1];
    const float* in_b  = (const float*)d_in[2];
    auto LP = [&](int l, int k) -> const float* { return (const float*)d_in[3 + l * 23 + k]; };
    const int FB = 3 + N_LAYER * 23;  // 49
    const float* lnf_s  = (const float*)d_in[FB + 0];
    const float* lnf_b  = (const float*)d_in[FB + 1];
    const float* mask_w = (const float*)d_in[FB + 2];
    const float* mask_b = (const float*)d_in[FB + 3];
    const float* out_w  = (const float*)d_in[FB + 4];
    const float* out_b  = (const float*)d_in[FB + 5];

    Bump ws{(char*)d_ws, 0, ws_size};
    float*    x    = (float*)ws.alloc(sizeof(float) * BT * N_EMBD);
    float*    xln  = (float*)ws.alloc(sizeof(float) * BT * N_EMBD);
    _Float16* mixh = (_Float16*)ws.alloc(sizeof(_Float16) * BT * N_EMBD);
    float*    pr[6];
    for (int i = 0; i < 6; ++i) pr[i] = (float*)ws.alloc(sizeof(float) * BT * N_EMBD);
    float*    ahb  = (float*)ws.alloc(sizeof(float) * BT * N_EMBD);
    float*    bhb  = (float*)ws.alloc(sizeof(float) * BT * N_EMBD);
    float*    yb   = (float*)ws.alloc(sizeof(float) * BT * N_EMBD);
    _Float16* hid  = (_Float16*)ws.alloc(sizeof(_Float16) * BT * D4);
    float*    mskb = (float*)ws.alloc(sizeof(float) * BT * NUM_SRC * N_EMBD);
    _Float16* wt   = (_Float16*)ws.alloc(sizeof(_Float16) * (size_t)D4 * N_EMBD);

    const int TPB = 256;
    const int elems = BT * N_EMBD;

    k_embed<<<cdiv(elems, TPB), TPB, 0, stream>>>(mixin, in_w, in_b, x);

    for (int l = 0; l < N_LAYER; ++l) {
        const float* mu[6] = {LP(l,0), LP(l,1), LP(l,2), LP(l,3), LP(l,4), LP(l,5)};
        const float* W[6]  = {LP(l,6), LP(l,7), LP(l,8), LP(l,9), LP(l,10), LP(l,11)};
        const float* w_o = LP(l,12);  const float* w_bias = LP(l,13);
        const float *ln1s = LP(l,14), *ln1b = LP(l,15);
        const float *ln2s = LP(l,16), *ln2b = LP(l,17);
        const float *gns = LP(l,18),  *gnb = LP(l,19), *mu_c = LP(l,20);
        const float *w_ck = LP(l,21), *w_cv = LP(l,22);

        // ---- time mix ----
        k_layernorm<<<BT / 8, 256, 0, stream>>>(x, ln1s, ln1b, xln, (_Float16*)nullptr);
        for (int p = 0; p < 6; ++p) {
            k_mix<<<cdiv(elems, TPB), TPB, 0, stream>>>(xln, mu[p], mixh);
            k_transpose<<<cdiv(N_EMBD * N_EMBD, TPB), TPB, 0, stream>>>(W[p], wt, N_EMBD, N_EMBD);
            int ep = (p == 1) ? EP_DECAY : (p >= 4 ? EP_SIGMOID : EP_F32);
            const float* bias = (p == 1) ? w_bias : nullptr;
            int tiles = (BT / 16) * (N_EMBD / 64);
            k_wmma_gemm<<<cdiv(tiles, 8), 256, 0, stream>>>(
                mixh, wt, pr[p], (_Float16*)nullptr, bias, BT, N_EMBD, N_EMBD, ep);
        }
        k_kkprep<<<cdiv(BT * N_HEAD, TPB), TPB, 0, stream>>>(pr[2], pr[4], ahb, bhb);
        k_scan<<<BATCH * N_HEAD, 64, 0, stream>>>(pr[0], pr[1], pr[2], pr[3], ahb, bhb, yb);
        k_gngate<<<cdiv(BT * N_HEAD, TPB), TPB, 0, stream>>>(yb, pr[5], gns, gnb, mixh);
        k_transpose<<<cdiv(N_EMBD * N_EMBD, TPB), TPB, 0, stream>>>(w_o, wt, N_EMBD, N_EMBD);
        {
            int tiles = (BT / 16) * (N_EMBD / 64);
            k_wmma_gemm<<<cdiv(tiles, 8), 256, 0, stream>>>(
                mixh, wt, x, (_Float16*)nullptr, nullptr, BT, N_EMBD, N_EMBD, EP_ADD);
        }

        // ---- channel mix ----
        k_layernorm<<<BT / 8, 256, 0, stream>>>(x, ln2s, ln2b, xln, (_Float16*)nullptr);
        k_mix<<<cdiv(elems, TPB), TPB, 0, stream>>>(xln, mu_c, mixh);
        k_transpose<<<cdiv(N_EMBD * D4, TPB), TPB, 0, stream>>>(w_ck, wt, N_EMBD, D4);
        {
            int tiles = (BT / 16) * (D4 / 64);
            k_wmma_gemm<<<cdiv(tiles, 8), 256, 0, stream>>>(
                mixh, wt, (float*)nullptr, hid, nullptr, BT, D4, N_EMBD, EP_RELU2_H);
        }
        k_transpose<<<cdiv(D4 * N_EMBD, TPB), TPB, 0, stream>>>(w_cv, wt, D4, N_EMBD);
        {
            int tiles = (BT / 16) * (N_EMBD / 64);
            k_wmma_gemm<<<cdiv(tiles, 8), 256, 0, stream>>>(
                hid, wt, x, (_Float16*)nullptr, nullptr, BT, N_EMBD, D4, EP_ADD);
        }
    }

    // ---- head: lnf -> sigmoid mask GEMM -> weighted reduce ----
    k_layernorm<<<BT / 8, 256, 0, stream>>>(x, lnf_s, lnf_b, xln, mixh);
    k_transpose<<<cdiv(N_EMBD * NUM_SRC * N_EMBD, TPB), TPB, 0, stream>>>(
        mask_w, wt, N_EMBD, NUM_SRC * N_EMBD);
    {
        int tiles = (BT / 16) * ((NUM_SRC * N_EMBD) / 64);
        k_wmma_gemm<<<cdiv(tiles, 8), 256, 0, stream>>>(
            mixh, wt, mskb, (_Float16*)nullptr, mask_b, BT, NUM_SRC * N_EMBD, N_EMBD, EP_SIGMOID);
    }
    k_maskout<<<cdiv(BT * NUM_SRC, TPB), TPB, 0, stream>>>(mskb, xln, out_w, out_b, (float*)d_out);
}
